// AdaptiveChannelGroupingMechanism_61495341744497
// MI455X (gfx1250) — compile-verified, matrix operand order checked by reading, not run on hardware
//
#include <hip/hip_runtime.h>
#include <math.h>

typedef __attribute__((ext_vector_type(16))) _Float16 v16h;
typedef __attribute__((ext_vector_type(8)))  _Float16 v8h;
typedef __attribute__((ext_vector_type(8)))  float    v8f;

#define NB 16
#define NC 256
#define NH 56
#define NW 56
#define HW 3136
#define NG 8
#define GS 32
#define CG 2048
#define BGN 128
#define BN_INV 0.9999950000374997f

union AV { v16h v; v8h h[2]; };

// K index inside a 32-wide chunk for gfx1250 16-bit WMMA A/B per-lane layout:
// lanes<16 hold K{0..7,16..23}; lanes>=16 hold K{8..15,24..31}
__device__ __forceinline__ int kofs(int half, int j) {
  return ((j < 8) ? j : (j - 8) + 16) + (half ? 8 : 0);
}

__device__ __forceinline__ v8f wmma_f16(v16h a, v16h b, v8f c) {
  return __builtin_amdgcn_wmma_f32_16x16x32_f16(false, a, false, b, (short)0, c, false, false);
}

__device__ __forceinline__ float sigm(float x){ return 1.f/(1.f+expf(-x)); }

// ---------------- stats: gap_x and per-channel L2 norm ----------------
__global__ void k_gap(const float* __restrict__ x, float* gap, float* xn) {
  int bc = blockIdx.x;
  const float* p = x + (size_t)bc * HW;
  float s = 0.f, s2 = 0.f;
  for (int i = threadIdx.x; i < HW; i += 256) { float v = p[i]; s += v; s2 += v*v; }
  __shared__ float r1[256], r2[256];
  r1[threadIdx.x] = s; r2[threadIdx.x] = s2; __syncthreads();
  for (int off = 128; off > 0; off >>= 1) {
    if (threadIdx.x < off) { r1[threadIdx.x] += r1[threadIdx.x+off]; r2[threadIdx.x] += r2[threadIdx.x+off]; }
    __syncthreads();
  }
  if (threadIdx.x == 0) { gap[bc] = r1[0]/(float)HW; xn[bc] = sqrtf(r2[0]); }
}

// ---------------- weight packing to WMMA B layout (f16) ----------------
__global__ void k_pack_rp(const float* __restrict__ w, _Float16* out) { // rp_cw (64,256)
  int idx = blockIdx.x*256 + threadIdx.x;
  if (idx >= 4*8*32*16) return;
  int j = idx & 15, lane = (idx>>4)&31, kc = (idx>>9)&7, nt = idx>>12;
  int n = nt*16 + (lane & 15);
  int K = kc*32 + kofs(lane>=16, j);
  out[idx] = (_Float16)w[n*256 + K];
}
__global__ void k_pack_se1(const float* __restrict__ w, _Float16* out) { // se1_w (32,32,3,3)
  int idx = blockIdx.x*256 + threadIdx.x;
  if (idx >= 9*2*32*16) return;
  int j = idx & 15, lane=(idx>>4)&31, nt=(idx>>9)&1, tap=idx>>10;
  int o = nt*16 + (lane&15);
  int i = kofs(lane>=16, j);
  out[idx] = (_Float16)w[(o*32+i)*9 + tap];
}
__global__ void k_pack_se2(const float* __restrict__ w, _Float16* out) { // se2_w (32,32)
  int idx = blockIdx.x*256 + threadIdx.x;
  if (idx >= 2*32*16) return;
  int j = idx & 15, lane=(idx>>4)&31, nt=idx>>9;
  int o = nt*16 + (lane&15);
  int i = kofs(lane>=16, j);
  out[idx] = (_Float16)w[o*32+i];
}

// ------- rp 1x1 conv via WMMA: rh_sum[b,o] = sum_p relu(x@rp_cw^T + b) -------
__global__ void __launch_bounds__(32) k_rp_gemm(const float* __restrict__ x,
                          const _Float16* __restrict__ rpp,
                          const float* __restrict__ rp_cb, float* __restrict__ rh) {
  int tile = blockIdx.x;            // 196 tiles of 16 pixels
  int b = blockIdx.y;
  int lane = threadIdx.x;
  int half = lane >= 16;
  int p = tile*16 + (lane & 15);
  v8f acc[4] = {};
  const v16h* bmat = (const v16h*)rpp;
  for (int kc = 0; kc < 8; ++kc) {
    AV a;
#pragma unroll
    for (int j = 0; j < 16; ++j) {
      int K = kc*32 + kofs(half, j);
      a.v[j] = (_Float16)x[((size_t)b*NC + K)*HW + p];
    }
#pragma unroll
    for (int nt = 0; nt < 4; ++nt)
      acc[nt] = wmma_f16(a.v, bmat[(nt*8+kc)*32 + lane], acc[nt]);
  }
#pragma unroll
  for (int nt = 0; nt < 4; ++nt) {
    int o = nt*16 + (lane & 15);
    float bia = rp_cb[o];
    float s = 0.f;
#pragma unroll
    for (int r = 0; r < 8; ++r) { float v = acc[nt][r] + bia; s += (v > 0.f ? v : 0.f); }
    atomicAdd(&rh[b*64 + o], s);
  }
}

// ---------------- content-aware feature cf (tiny MLPs) ----------------
__global__ void k_cf(const float* gap, const float* w1, const float* b1,
                     const float* w2, const float* b2,
                     const float* w3, const float* b3, float* cf) {
  int b = blockIdx.x; int t = threadIdx.x;
  __shared__ float h1[16], h2[16];
  if (t < 16) {
    float s = b1[t];
    for (int k = 0; k < 256; ++k) s += gap[b*256+k]*w1[t*256+k];
    h1[t] = s > 0.f ? s : 0.f;
  }
  __syncthreads();
  if (t < 16) {
    float s = b2[t];
    for (int k = 0; k < 16; ++k) s += h1[k]*w2[t*16+k];
    h2[t] = s > 0.f ? s : 0.f;
  }
  __syncthreads();
  for (int o = t; o < CG; o += 256) {
    float s = b3[o];
    for (int k = 0; k < 16; ++k) s += h2[k]*w3[o*16+k];
    cf[(size_t)b*CG+o] = s;
  }
}

__global__ void k_sgh(const float* cf, const float* w, const float* bia, float* sgh) {
  int b = blockIdx.x, o = threadIdx.x;
  float s = bia[o];
  const float* cfe = cf + (size_t)b*CG;
  const float* wr  = w  + (size_t)o*CG;
  for (int k = 0; k < CG; ++k) s += cfe[k]*wr[k];
  sgh[b*256+o] = s > 0.f ? s : 0.f;
}

__global__ void k_sgout(const float* sgh, const float* w, const float* bia, float* out) {
  int b = blockIdx.x; int o = blockIdx.y*256 + threadIdx.x;
  float s = bia[o];
  const float* hr = sgh + b*256;
  const float* wr = w + (size_t)o*256;
  for (int k = 0; k < 256; ++k) s += hr[k]*wr[k];
  out[(size_t)b*CG + o] = s;
}

__global__ void k_softmax2048(float* v) {
  int b = blockIdx.x; float* row = v + (size_t)b*CG;
  __shared__ float red[256];
  float mx = -1e30f;
  for (int i = threadIdx.x; i < CG; i += 256) mx = fmaxf(mx, row[i]);
  red[threadIdx.x] = mx; __syncthreads();
  for (int off = 128; off > 0; off >>= 1) {
    if (threadIdx.x < off) red[threadIdx.x] = fmaxf(red[threadIdx.x], red[threadIdx.x+off]);
    __syncthreads();
  }
  float m = red[0]; __syncthreads();
  float s = 0.f;
  for (int i = threadIdx.x; i < CG; i += 256) s += expf(row[i]-m);
  red[threadIdx.x] = s; __syncthreads();
  for (int off = 128; off > 0; off >>= 1) {
    if (threadIdx.x < off) red[threadIdx.x] += red[threadIdx.x+off];
    __syncthreads();
  }
  float inv = 1.f/red[0]; __syncthreads();
  for (int i = threadIdx.x; i < CG; i += 256) row[i] = expf(row[i]-m)*inv;
}

__global__ void k_conf(const float* cf, const float* w1, const float* b1,
                       const float* w2, const float* b2, float* conf) {
  int b = blockIdx.x, t = threadIdx.x;
  __shared__ float h[64];
  float s = b1[t];
  const float* cfe = cf + (size_t)b*CG;
  const float* wr = w1 + (size_t)t*CG;
  for (int k = 0; k < CG; ++k) s += cfe[k]*wr[k];
  h[t] = s > 0.f ? s : 0.f;
  __syncthreads();
  if (t == 0) {
    float v = b2[0];
    for (int k = 0; k < 64; ++k) v += h[k]*w2[k];
    conf[b] = sigm(v);
  }
}

__global__ void k_logits0(const float* rh, const float* w, const float* bia,
                          const float* ga, float* logits) {
  int b = blockIdx.x; int o = blockIdx.y*256 + threadIdx.x;
  float s = bia[o];
  const float* wr = w + (size_t)o*64;
  for (int k = 0; k < 64; ++k) s += (rh[b*64+k]*(1.f/HW))*wr[k];
  logits[(size_t)b*CG+o] = s + logf(ga[(size_t)b*CG+o] + 1e-8f);
}

// ---------------- router ----------------
__global__ void k_rw_softmax8(const float* logits, float* rw) {
  int b = blockIdx.x, c = threadIdx.x;
  const float* l = logits + (size_t)b*CG + c*NG;
  float mx = -1e30f; float e[NG];
  for (int g = 0; g < NG; ++g) mx = fmaxf(mx, l[g]);
  float s = 0.f;
  for (int g = 0; g < NG; ++g) { e[g] = expf(l[g]-mx); s += e[g]; }
  float inv = 1.f/s;
  float* o = rw + (size_t)b*CG + c*NG;
  for (int g = 0; g < NG; ++g) o[g] = e[g]*inv;
}

__global__ void k_go(const float* __restrict__ x, const float* __restrict__ rw,
                     float* __restrict__ go, float* __restrict__ gn_sq) {
  int b = blockIdx.y;
  int p = blockIdx.x*256 + threadIdx.x;
  __shared__ float w[CG];
  for (int i = threadIdx.x; i < CG; i += 256) w[i] = rw[(size_t)b*CG+i];
  __syncthreads();
  float acc[NG] = {};
  if (p < HW) {
    for (int c = 0; c < NC; ++c) {
      float xv = x[((size_t)b*NC + c)*HW + p];
#pragma unroll
      for (int g = 0; g < NG; ++g) acc[g] += w[c*NG+g]*xv;
    }
    for (int g = 0; g < NG; ++g) go[((size_t)b*NG+g)*HW + p] = acc[g];
  }
  __shared__ float red[256];
  for (int g = 0; g < NG; ++g) {
    red[threadIdx.x] = (p < HW) ? acc[g]*acc[g] : 0.f;
    __syncthreads();
    for (int off = 128; off > 0; off >>= 1) {
      if (threadIdx.x < off) red[threadIdx.x] += red[threadIdx.x+off];
      __syncthreads();
    }
    if (threadIdx.x == 0) atomicAdd(&gn_sq[b*NG+g], red[0]);
    __syncthreads();
  }
}

__global__ void k_dots_update(const float* __restrict__ x, const float* __restrict__ go,
                              const float* __restrict__ xn, const float* __restrict__ gn_sq,
                              float* __restrict__ logits) {
  int bc = blockIdx.x; int b = bc >> 8; int c = bc & 255;
  float acc[NG] = {};
  for (int p = threadIdx.x; p < HW; p += 256) {
    float xv = x[(size_t)bc*HW + p];
#pragma unroll
    for (int g = 0; g < NG; ++g) acc[g] += xv * go[((size_t)b*NG+g)*HW + p];
  }
  __shared__ float red[256];
  __shared__ float dots[NG];
  for (int g = 0; g < NG; ++g) {
    red[threadIdx.x] = acc[g]; __syncthreads();
    for (int off = 128; off > 0; off >>= 1) {
      if (threadIdx.x < off) red[threadIdx.x] += red[threadIdx.x+off];
      __syncthreads();
    }
    if (threadIdx.x == 0) dots[g] = red[0];
    __syncthreads();
  }
  if (threadIdx.x < NG) {
    int g = threadIdx.x;
    float xnv = fmaxf(xn[bc], 1e-8f);
    float gnv = fmaxf(sqrtf(gn_sq[b*NG+g]), 1e-8f);
    logits[(size_t)b*CG + c*NG + g] += dots[g]/(xnv*gnv);
  }
}

// ---------------- group split xg = x * wsel ----------------
__global__ void k_xg(const float* __restrict__ x, const float* __restrict__ fwv,
                     float* __restrict__ xg, float* __restrict__ xg_gap) {
  int bc = blockIdx.x; int b = bc>>8; int c = bc&255; int g = c>>5; int cl = c&31;
  int p = blockIdx.y*256 + threadIdx.x;
  float w = fwv[(size_t)b*CG + c*NG + g];
  float v = 0.f;
  if (p < HW) { v = x[(size_t)bc*HW+p]*w; xg[(size_t)bc*HW+p] = v; }
  __shared__ float red[256];
  red[threadIdx.x] = v; __syncthreads();
  for (int off = 128; off > 0; off >>= 1) {
    if (threadIdx.x < off) red[threadIdx.x] += red[threadIdx.x+off];
    __syncthreads();
  }
  if (threadIdx.x == 0) atomicAdd(&xg_gap[(b*NG+g)*GS+cl], red[0]);
}

__global__ void k_imp(const float* xg_gap, const float* w1, const float* b1,
                      const float* w2, const float* b2, float* imp) {
  int bg = blockIdx.x, t = threadIdx.x;
  __shared__ float m[GS], hid[8];
  m[t] = xg_gap[bg*GS+t]*(1.f/HW);
  __syncthreads();
  if (t < 8) {
    float s = b1[t]; for (int k = 0; k < GS; ++k) s += m[k]*w1[t*GS+k];
    hid[t] = s > 0.f ? s : 0.f;
  }
  __syncthreads();
  float s = b2[t]; for (int k = 0; k < 8; ++k) s += hid[k]*w2[t*8+k];
  imp[bg*GS+t] = sigm(s);
}

// --- depthwise chain -> "weighted" staged as NHWC f16 for the WMMA conv ---
__global__ void k_dwconv(const float* __restrict__ xg,
                         const float* ci1w, const float* ci1b,
                         const float* bn1s, const float* bn1b,
                         const float* ci2w, const float* ci2b,
                         const float* bn2s, const float* bn2b,
                         const float* __restrict__ imp,
                         _Float16* __restrict__ wgt) {
  int bg = blockIdx.y;
  int tile = blockIdx.x; int tx = tile & 3, ty = tile >> 2;
  int x0 = tx*16, y0 = ty*16;
  int lx = threadIdx.x & 15, ly = threadIdx.x >> 4;
  __shared__ float hs[18*18];
  for (int cl = 0; cl < GS; ++cl) {
    float a1 = ci1w[cl], c1 = ci1b[cl];
    float s1 = bn1s[cl]*BN_INV, o1 = bn1b[cl];
    const float* plane = xg + ((size_t)bg*GS + cl)*HW;
    for (int idx = threadIdx.x; idx < 18*18; idx += 256) {
      int iy = idx/18, ix = idx%18;
      int gy = y0+iy-1, gx = x0+ix-1;
      float v = 0.f;
      if (gy >= 0 && gy < NH && gx >= 0 && gx < NW) {
        float t = plane[gy*NW+gx]*a1 + c1;
        t = t*s1 + o1;
        v = t > 0.f ? t : 0.f;
      }
      hs[idx] = v;
    }
    __syncthreads();
    int oy = y0+ly, ox = x0+lx;
    if (oy < NH && ox < NW) {
      float acc = ci2b[cl];
#pragma unroll
      for (int dy = 0; dy < 3; ++dy)
#pragma unroll
        for (int dx = 0; dx < 3; ++dx)
          acc += ci2w[cl*9+dy*3+dx]*hs[(ly+dy)*18 + (lx+dx)];
      float v = acc*(bn2s[cl]*BN_INV)+bn2b[cl];
      v = v > 0.f ? v : 0.f;
      v *= imp[bg*GS+cl];
      wgt[((size_t)bg*HW + oy*NW+ox)*GS + cl] = (_Float16)v;
    }
    __syncthreads();
  }
}

// ---- se1: 3x3 dense 32->32 conv as implicit GEMM (9 taps x 2 WMMA tiles) ----
__global__ void __launch_bounds__(32) k_se1(const _Float16* __restrict__ wgt,
                      const _Float16* __restrict__ se1p,
                      const float* se1b, const float* bn3s, const float* bn3b,
                      _Float16* __restrict__ sact) {
  int tile = blockIdx.x;   // 196 tiles of 16 pixels
  int bg = blockIdx.y;
  int lane = threadIdx.x;
  int half = lane >= 16;
  int p = tile*16 + (lane & 15);
  int y = p/NW, x = p%NW;
  v8f acc[2] = {};
  const v16h* bm = (const v16h*)se1p;
  for (int t = 0; t < 9; ++t) {
    int ny = y + t/3 - 1, nx = x + t%3 - 1;
    AV a;
    if (ny >= 0 && ny < NH && nx >= 0 && nx < NW) {
      const v8h* ap = (const v8h*)(wgt + ((size_t)bg*HW + ny*NW+nx)*GS);
      a.h[0] = ap[half];
      a.h[1] = ap[2+half];
    } else {
#pragma unroll
      for (int j = 0; j < 16; ++j) a.v[j] = (_Float16)0.f;
    }
#pragma unroll
    for (int nt = 0; nt < 2; ++nt)
      acc[nt] = wmma_f16(a.v, bm[(t*2+nt)*32 + lane], acc[nt]);
  }
#pragma unroll
  for (int nt = 0; nt < 2; ++nt) {
    int n = nt*16 + (lane & 15);
    float sb = se1b[n], s3 = bn3s[n]*BN_INV, o3 = bn3b[n];
#pragma unroll
    for (int r = 0; r < 8; ++r) {
      int pout = tile*16 + r + (half ? 8 : 0);
      float v = acc[nt][r] + sb;
      v = v*s3 + o3; v = v > 0.f ? v : 0.f;
      sact[((size_t)bg*HW + pout)*GS + n] = (_Float16)v;
    }
  }
}

// ---- se2: 1x1 dense 32->32 conv via WMMA + bn4 + residual xg -> enh, tokens ----
__global__ void __launch_bounds__(32) k_se2(const _Float16* __restrict__ sact,
                      const _Float16* __restrict__ se2p,
                      const float* se2b, const float* bn4s, const float* bn4b,
                      const float* __restrict__ xg,
                      float* __restrict__ enh, float* __restrict__ tok) {
  int tile = blockIdx.x; int bg = blockIdx.y;
  int b = bg>>3, g = bg&7;
  int lane = threadIdx.x; int half = lane >= 16;
  int p = tile*16 + (lane & 15);
  AV a;
  const v8h* ap = (const v8h*)(sact + ((size_t)bg*HW + p)*GS);
  a.h[0] = ap[half];
  a.h[1] = ap[2+half];
  const v16h* bm = (const v16h*)se2p;
  v8f acc[2] = {};
#pragma unroll
  for (int nt = 0; nt < 2; ++nt) acc[nt] = wmma_f16(a.v, bm[nt*32+lane], acc[nt]);
#pragma unroll
  for (int nt = 0; nt < 2; ++nt) {
    int n = nt*16 + (lane & 15);
    float sb = se2b[n], s4 = bn4s[n]*BN_INV, o4 = bn4b[n];
    size_t chBase = ((size_t)b*NC + g*GS + n)*HW;
    float tsum = 0.f;
#pragma unroll
    for (int r = 0; r < 8; ++r) {
      int pout = tile*16 + r + (half ? 8 : 0);
      float e = acc[nt][r] + sb;
      e = e*s4 + o4 + xg[chBase + pout];
      enh[chBase + pout] = e;
      tsum += e;
    }
    atomicAdd(&tok[bg*GS+n], tsum);
  }
}

// ---------------- MHA over 8 group tokens + group fusion -> fw ----------------
__global__ void k_mha_gf(const float* tok,
                         const float* inw, const float* inb,
                         const float* outw, const float* outb,
                         const float* gw1, const float* gb1,
                         const float* gw2, const float* gb2,
                         float* fw) {
  int b = blockIdx.x, t = threadIdx.x;
  __shared__ float tk[NG][GS];
  __shared__ float qkv[NG][96];
  __shared__ float sc[NG][NG];
  __shared__ float av[NG][GS];
  __shared__ float ao[NG*GS];
  __shared__ float gh[128];
  for (int i = t; i < NG*GS; i += 256) tk[i/GS][i%GS] = tok[(size_t)b*NG*GS + i]*(1.f/HW);
  __syncthreads();
  for (int i = t; i < NG*96; i += 256) {
    int g = i/96, j = i%96;
    float s = inb[j];
    for (int k = 0; k < GS; ++k) s += tk[g][k]*inw[j*GS+k];
    qkv[g][j] = s;
  }
  __syncthreads();
  if (t < NG*NG) {
    int q = t/NG, k = t%NG;
    float s = 0.f;
    for (int d = 0; d < GS; ++d) s += qkv[q][d]*qkv[k][GS+d];
    sc[q][k] = s*(1.f/sqrtf((float)GS));
  }
  __syncthreads();
  if (t < NG) {
    float mx = -1e30f; for (int k = 0; k < NG; ++k) mx = fmaxf(mx, sc[t][k]);
    float s = 0.f; for (int k = 0; k < NG; ++k) { sc[t][k] = expf(sc[t][k]-mx); s += sc[t][k]; }
    float inv = 1.f/s; for (int k = 0; k < NG; ++k) sc[t][k] *= inv;
  }
  __syncthreads();
  if (t < NG*GS) {
    int q = t/GS, d = t%GS;
    float s = 0.f;
    for (int k = 0; k < NG; ++k) s += sc[q][k]*qkv[k][2*GS+d];
    av[q][d] = s;
  }
  __syncthreads();
  if (t < NG*GS) {
    int q = t/GS, d = t%GS;
    float s = outb[d];
    for (int k = 0; k < GS; ++k) s += av[q][k]*outw[d*GS+k];
    ao[q*GS+d] = s;
  }
  __syncthreads();
  if (t < 128) {
    float s = gb1[t];
    for (int k = 0; k < NG*GS; ++k) s += ao[k]*gw1[t*NG*GS+k];
    gh[t] = s > 0.f ? s : 0.f;
  }
  __syncthreads();
  {
    float s = gb2[t];
    for (int k = 0; k < 128; ++k) s += gh[k]*gw2[t*128+k];
    fw[(size_t)b*NC + t] = sigm(s);
  }
}

// ---------------- adaptive fusion gates ----------------
__global__ void k_gates(const float* gap, const float* tok, const float* fw,
                        const float* conf,
                        const float* fgw1, const float* fgb1, const float* fgw2, const float* fgb2,
                        const float* cgw1, const float* cgb1, const float* cgw2, const float* cgb2,
                        const float* cmw1, const float* cmb1, const float* cmw2, const float* cmb2,
                        float* gA, float* oma) {
  int b = blockIdx.x, t = threadIdx.x;
  __shared__ float comb[2*NC];
  __shared__ float h64[64];
  __shared__ float h128[128];
  __shared__ float sgv, aSh;
  comb[t]    = gap[b*NC+t];
  comb[NC+t] = tok[(size_t)b*NC + t]*(1.f/HW)*fw[(size_t)b*NC+t];  // enhanced.mean
  __syncthreads();
  if (t < 64) {
    float s = fgb1[t];
    for (int k = 0; k < 2*NC; ++k) s += comb[k]*fgw1[t*2*NC+k];
    h64[t] = s > 0.f ? s : 0.f;
  }
  if (t < 128) {
    float s = cgb1[t];
    for (int k = 0; k < 2*NC; ++k) s += comb[k]*cgw1[t*2*NC+k];
    h128[t] = s > 0.f ? s : 0.f;
  }
  __syncthreads();
  if (t == 0) {
    float s = fgb2[0];
    for (int k = 0; k < 64; ++k) s += h64[k]*fgw2[k];
    sgv = sigm(s);
  }
  float cgate;
  {
    float s = cgb2[t];
    for (int k = 0; k < 128; ++k) s += h128[k]*cgw2[t*128+k];
    cgate = sigm(s);
  }
  __syncthreads();
  if (t < 64) {
    float s = cmw1[t]*conf[b] + cmb1[t];
    h64[t] = s > 0.f ? s : 0.f;
  }
  __syncthreads();
  if (t == 0) {
    float s = cmb2[0];
    for (int k = 0; k < 64; ++k) s += h64[k]*cmw2[k];
    aSh = sigm(s)*sgv;
  }
  __syncthreads();
  gA[(size_t)b*NC+t] = aSh*cgate*fw[(size_t)b*NC+t];  // fold fw into the gate
  if (t == 0) oma[b] = 1.f - aSh;
}

// ---------------- final blend ----------------
__global__ void k_final(const float* __restrict__ enh, const float* __restrict__ x,
                        const float* __restrict__ gA, const float* __restrict__ oma,
                        float* __restrict__ out) {
  int bc = blockIdx.x; int b = bc>>8;
  int p = blockIdx.y*256 + threadIdx.x;
  if (p >= HW) return;
  size_t idx = (size_t)bc*HW + p;
  out[idx] = gA[bc]*enh[idx] + oma[b]*x[idx];
}

extern "C" void kernel_launch(void* const* d_in, const int* in_sizes, int n_in,
                              void* d_out, int out_size, void* d_ws, size_t ws_size,
                              hipStream_t stream) {
  (void)in_sizes; (void)n_in; (void)out_size; (void)ws_size;
  const float* x     = (const float*)d_in[0];
  const float* ca_w1 = (const float*)d_in[1];  const float* ca_b1 = (const float*)d_in[2];
  const float* ca_w2 = (const float*)d_in[3];  const float* ca_b2 = (const float*)d_in[4];
  const float* ca_w3 = (const float*)d_in[5];  const float* ca_b3 = (const float*)d_in[6];
  const float* sg_w1 = (const float*)d_in[7];  const float* sg_b1 = (const float*)d_in[8];
  const float* sg_w2 = (const float*)d_in[9];  const float* sg_b2 = (const float*)d_in[10];
  const float* ce_w1 = (const float*)d_in[11]; const float* ce_b1 = (const float*)d_in[12];
  const float* ce_w2 = (const float*)d_in[13]; const float* ce_b2 = (const float*)d_in[14];
  const float* rp_cw = (const float*)d_in[15]; const float* rp_cb = (const float*)d_in[16];
  const float* rp_lw = (const float*)d_in[17]; const float* rp_lb = (const float*)d_in[18];
  const float* ci1_w = (const float*)d_in[19]; const float* ci1_b = (const float*)d_in[20];
  const float* bn1_s = (const float*)d_in[21]; const float* bn1_b = (const float*)d_in[22];
  const float* ci2_w = (const float*)d_in[23]; const float* ci2_b = (const float*)d_in[24];
  const float* bn2_s = (const float*)d_in[25]; const float* bn2_b = (const float*)d_in[26];
  const float* imp_w1 = (const float*)d_in[27]; const float* imp_b1 = (const float*)d_in[28];
  const float* imp_w2 = (const float*)d_in[29]; const float* imp_b2 = (const float*)d_in[30];
  const float* se1_w = (const float*)d_in[31]; const float* se1_b = (const float*)d_in[32];
  const float* bn3_s = (const float*)d_in[33]; const float* bn3_b = (const float*)d_in[34];
  const float* se2_w = (const float*)d_in[35]; const float* se2_b = (const float*)d_in[36];
  const float* bn4_s = (const float*)d_in[37]; const float* bn4_b = (const float*)d_in[38];
  const float* mha_in_w  = (const float*)d_in[39]; const float* mha_in_b  = (const float*)d_in[40];
  const float* mha_out_w = (const float*)d_in[41]; const float* mha_out_b = (const float*)d_in[42];
  const float* gf_w1 = (const float*)d_in[43]; const float* gf_b1 = (const float*)d_in[44];
  const float* gf_w2 = (const float*)d_in[45]; const float* gf_b2 = (const float*)d_in[46];
  const float* fg_w1 = (const float*)d_in[47]; const float* fg_b1 = (const float*)d_in[48];
  const float* fg_w2 = (const float*)d_in[49]; const float* fg_b2 = (const float*)d_in[50];
  const float* cg_w1 = (const float*)d_in[51]; const float* cg_b1 = (const float*)d_in[52];
  const float* cg_w2 = (const float*)d_in[53]; const float* cg_b2 = (const float*)d_in[54];
  const float* cm_w1 = (const float*)d_in[55]; const float* cm_b1 = (const float*)d_in[56];
  const float* cm_w2 = (const float*)d_in[57]; const float* cm_b2 = (const float*)d_in[58];
  float* out = (float*)d_out;

  const size_t N = (size_t)NB*NC*HW;
  char* wsb = (char*)d_ws;
  size_t off = 0;
  auto alloc = [&](size_t bytes) -> void* {
    void* p = wsb + off;
    off = (off + bytes + 255) & ~(size_t)255;
    return p;
  };
  float* gap    = (float*)alloc(NB*NC*4);
  float* xn     = (float*)alloc(NB*NC*4);
  float* rh     = (float*)alloc(NB*64*4);
  float* cf     = (float*)alloc((size_t)NB*CG*4);
  float* sgh    = (float*)alloc(NB*256*4);
  float* ga     = (float*)alloc((size_t)NB*CG*4);
  float* conf   = (float*)alloc(NB*4);
  float* logits = (float*)alloc((size_t)NB*CG*4);
  float* rw     = (float*)alloc((size_t)NB*CG*4);
  float* go     = (float*)alloc((size_t)NB*NG*HW*4);
  float* gn_sq  = (float*)alloc(NB*NG*4);
  float* xg     = (float*)alloc(N*4);
  float* xg_gap = (float*)alloc(BGN*GS*4);
  float* imp    = (float*)alloc(BGN*GS*4);
  float* tok    = (float*)alloc(BGN*GS*4);
  float* fw     = (float*)alloc(NB*NC*4);
  float* gA     = (float*)alloc(NB*NC*4);
  float* oma    = (float*)alloc(NB*4);
  _Float16* rpp  = (_Float16*)alloc(4*8*32*16*2);
  _Float16* se1p = (_Float16*)alloc(9*2*32*16*2);
  _Float16* se2p = (_Float16*)alloc(2*32*16*2);
  _Float16* wgt  = (_Float16*)alloc((size_t)BGN*HW*GS*2);
  _Float16* sact = (_Float16*)alloc((size_t)BGN*HW*GS*2);
  float* enh    = (float*)alloc(N*4);

  hipMemsetAsync(rh, 0, NB*64*4, stream);
  k_gap<<<NB*NC, 256, 0, stream>>>(x, gap, xn);
  k_pack_rp<<<64, 256, 0, stream>>>(rp_cw, rpp);
  k_pack_se1<<<36, 256, 0, stream>>>(se1_w, se1p);
  k_pack_se2<<<4, 256, 0, stream>>>(se2_w, se2p);
  k_rp_gemm<<<dim3(196, NB), 32, 0, stream>>>(x, rpp, rp_cb, rh);
  k_cf<<<NB, 256, 0, stream>>>(gap, ca_w1, ca_b1, ca_w2, ca_b2, ca_w3, ca_b3, cf);
  k_sgh<<<NB, 256, 0, stream>>>(cf, sg_w1, sg_b1, sgh);
  k_sgout<<<dim3(NB, 8), 256, 0, stream>>>(sgh, sg_w2, sg_b2, ga);
  k_softmax2048<<<NB, 256, 0, stream>>>(ga);
  k_conf<<<NB, 64, 0, stream>>>(cf, ce_w1, ce_b1, ce_w2, ce_b2, conf);
  k_logits0<<<dim3(NB, 8), 256, 0, stream>>>(rh, rp_lw, rp_lb, ga, logits);

  for (int it = 0; it < 2; ++it) {
    k_rw_softmax8<<<NB, 256, 0, stream>>>(logits, rw);
    hipMemsetAsync(gn_sq, 0, NB*NG*4, stream);
    k_go<<<dim3(13, NB), 256, 0, stream>>>(x, rw, go, gn_sq);
    k_dots_update<<<NB*NC, 256, 0, stream>>>(x, go, xn, gn_sq, logits);
  }
  k_rw_softmax8<<<NB, 256, 0, stream>>>(logits, rw);  // final_w

  hipMemsetAsync(xg_gap, 0, BGN*GS*4, stream);
  k_xg<<<dim3(NB*NC, 13), 256, 0, stream>>>(x, rw, xg, xg_gap);
  k_imp<<<BGN, 32, 0, stream>>>(xg_gap, imp_w1, imp_b1, imp_w2, imp_b2, imp);
  k_dwconv<<<dim3(16, BGN), 256, 0, stream>>>(xg, ci1_w, ci1_b, bn1_s, bn1_b,
                                              ci2_w, ci2_b, bn2_s, bn2_b, imp, wgt);
  k_se1<<<dim3(196, BGN), 32, 0, stream>>>(wgt, se1p, se1_b, bn3_s, bn3_b, sact);
  hipMemsetAsync(tok, 0, BGN*GS*4, stream);
  k_se2<<<dim3(196, BGN), 32, 0, stream>>>(sact, se2p, se2_b, bn4_s, bn4_b, xg, enh, tok);
  k_mha_gf<<<NB, 256, 0, stream>>>(tok, mha_in_w, mha_in_b, mha_out_w, mha_out_b,
                                   gf_w1, gf_b1, gf_w2, gf_b2, fw);
  k_gates<<<NB, 256, 0, stream>>>(gap, tok, fw, conf,
                                  fg_w1, fg_b1, fg_w2, fg_b2,
                                  cg_w1, cg_b1, cg_w2, cg_b2,
                                  cm_w1, cm_b1, cm_w2, cm_b2, gA, oma);
  k_final<<<dim3(NB*NC, 13), 256, 0, stream>>>(enh, x, gA, oma, out);
}
